// MHAttention_16140487098390
// MI455X (gfx1250) — matrix operand order repaired and checked by reading, not compiled, not run on hardware
//
#include <hip/hip_runtime.h>
#include <math.h>

typedef _Float16 f16;
typedef __attribute__((ext_vector_type(16))) _Float16 v16h;
typedef __attribute__((ext_vector_type(8)))  _Float16 v8h;
typedef __attribute__((ext_vector_type(4)))  _Float16 v4h;
typedef __attribute__((ext_vector_type(8)))  float    v8f;
typedef __attribute__((ext_vector_type(4)))  int      v4i;

#define Tdim 4096
#define Edim 1024
#define Hn   16
#define HDim 64
#define ATT_SCALE 0.125f   // 8 / 64

#if __has_builtin(__builtin_amdgcn_update_dpp)
#define HAVE_DPP 1
#endif
#if __has_builtin(__builtin_amdgcn_global_load_async_to_lds_b128) && \
    __has_builtin(__builtin_amdgcn_s_wait_asynccnt)
#define HAVE_ASYNC 1
#endif

#if defined(HAVE_ASYNC)
// async copy of 16 bytes global -> LDS, tracked by ASYNCcnt
static __device__ __forceinline__ void async_cp16(const f16* g, f16* l) {
  __builtin_amdgcn_global_load_async_to_lds_b128(
      (__attribute__((address_space(1))) v4i*)(void*)const_cast<f16*>(g),
      (__attribute__((address_space(3))) v4i*)(void*)l, 0, 0);
}
#endif
static __device__ __forceinline__ void cp16(const f16* g, f16* l) {
#if defined(HAVE_ASYNC)
  async_cp16(g, l);
#else
  *reinterpret_cast<v8h*>(l) = *reinterpret_cast<const v8h*>(g);
#endif
}
static __device__ __forceinline__ void wait_cp() {
#if defined(HAVE_ASYNC)
  __builtin_amdgcn_s_wait_asynccnt(0);
#endif
}

static __device__ __forceinline__ v16h cat8(v8h a, v8h b) {
  return __builtin_shufflevector(a, b, 0,1,2,3,4,5,6,7,8,9,10,11,12,13,14,15);
}

static __device__ __forceinline__ v8f wmma16(v16h a, v16h b, v8f c) {
  // D = A(16x32) * B(32x16) + C, f32 accumulate
  return __builtin_amdgcn_wmma_f32_16x16x32_f16(false, a, false, b, (short)0, c,
                                                false, false);
}

// Max-reduce across the 16 lanes of a DPP row (our half-wave): butterfly via
// quad_perm(xor1)=0xB1, quad_perm(xor2)=0x4E, row_half_mirror=0x141,
// row_mirror=0x140. VALU-only; no DS round-trips, no s_wait_dscnt.
static __device__ __forceinline__ float rowmax16(float x) {
#if defined(HAVE_DPP)
  int v, t;
  v = __float_as_int(x);
  t = __builtin_amdgcn_update_dpp(v, v, 0xB1, 0xF, 0xF, false);
  x = fmaxf(x, __int_as_float(t));
  v = __float_as_int(x);
  t = __builtin_amdgcn_update_dpp(v, v, 0x4E, 0xF, 0xF, false);
  x = fmaxf(x, __int_as_float(t));
  v = __float_as_int(x);
  t = __builtin_amdgcn_update_dpp(v, v, 0x141, 0xF, 0xF, false);
  x = fmaxf(x, __int_as_float(t));
  v = __float_as_int(x);
  t = __builtin_amdgcn_update_dpp(v, v, 0x140, 0xF, 0xF, false);
  x = fmaxf(x, __int_as_float(t));
  return x;
#else
#pragma unroll
  for (int mk = 8; mk >= 1; mk >>= 1) x = fmaxf(x, __shfl_xor(x, mk, 32));
  return x;
#endif
}

// Online-softmax update for one 16x32 score block sitting in (S0,S1).
// Writes exp(S - m_new) as f16 into this wave's private Ps region.
template <bool MASKED>
static __device__ __forceinline__ void softmax_step(
    const v8f& S0, const v8f& S1, int q0, int k0, int lo, int hi,
    float* mrow, float* lrow, v8f* Oacc, f16* Pw) {
#pragma unroll
  for (int r = 0; r < 8; ++r) {
    float s0 = S0[r] * ATT_SCALE;
    float s1 = S1[r] * ATT_SCALE;
    if (MASKED) {
      const int qi = q0 + r + 8 * hi;
      if (k0 + lo > qi)      s0 = -__builtin_inff();
      if (k0 + 16 + lo > qi) s1 = -__builtin_inff();
    }
    const float mx = rowmax16(fmaxf(s0, s1));
    const float mn = fmaxf(mrow[r], mx);
    const float alpha = __expf(mrow[r] - mn);
    mrow[r] = mn;
    lrow[r] *= alpha;                       // rowsum added via WMMA afterwards
    Pw[(r + 8 * hi) * 32 + lo]      = (f16)__expf(s0 - mn);
    Pw[(r + 8 * hi) * 32 + 16 + lo] = (f16)__expf(s1 - mn);
#pragma unroll
    for (int j = 0; j < 4; ++j) Oacc[j][r] *= alpha;
  }
}

// ---------------------------------------------------------------------------
// f32 -> f16 conversion, 4 elements per thread
// ---------------------------------------------------------------------------
__global__ void cvt_f32_f16(const float* __restrict__ s, f16* __restrict__ d, int n4) {
  int i = blockIdx.x * blockDim.x + threadIdx.x;
  if (i < n4) {
    float4 x = reinterpret_cast<const float4*>(s)[i];
    v4h h;
    h[0] = (f16)x.x; h[1] = (f16)x.y; h[2] = (f16)x.z; h[3] = (f16)x.w;
    reinterpret_cast<v4h*>(d)[i] = h;
  }
}

// ---------------------------------------------------------------------------
// Tiled WMMA GEMM: out[m,n] = sum_k X[m,k] * W[n,k]   (y = x @ W^T)
// Block tile 128x128, BK=32, double-buffered async LDS staging.
// MODE 0: f16 row-major; MODE 1: f16 transposed [N][M]; MODE 2: f32 row-major.
// ---------------------------------------------------------------------------
template <int MODE>
__global__ __launch_bounds__(256, 1)
void gemm_xwT(const f16* __restrict__ X, const f16* __restrict__ W,
              f16* __restrict__ outh, float* __restrict__ outf,
              int M, int N, int K) {
  __shared__ alignas(64) f16 As[2][128 * 32];
  __shared__ alignas(64) f16 Bs[2][128 * 32];

  const int tid  = threadIdx.x;
  const int lane = tid & 31;
  const int wid  = tid >> 5;
  const int lo   = lane & 15;
  const int hi   = lane >> 4;
  const int wm   = (wid >> 2) * 64;
  const int wn   = (wid & 3) * 32;
  const int m0   = blockIdx.y * 128;
  const int n0   = blockIdx.x * 128;

  auto stage = [&](int buf, int kk) {
#pragma unroll
    for (int it = 0; it < 2; ++it) {
      int id  = tid + it * 256;
      int row = id >> 2;
      int c   = id & 3;
      cp16(&X[(size_t)(m0 + row) * K + kk + c * 8], &As[buf][row * 32 + c * 8]);
      cp16(&W[(size_t)(n0 + row) * K + kk + c * 8], &Bs[buf][row * 32 + c * 8]);
    }
  };

  v8f z = {};
  v8f acc[8];
#pragma unroll
  for (int i = 0; i < 8; ++i) acc[i] = z;

  const int nk = K / 32;
  stage(0, 0);
  wait_cp();
  __syncthreads();

  for (int t = 0; t < nk; ++t) {
    const int cur = t & 1;
    if (t + 1 < nk) stage(cur ^ 1, (t + 1) * 32);   // prefetch next tile

    v16h a[4];
#pragma unroll
    for (int mi = 0; mi < 4; ++mi) {
      const f16* p = &As[cur][(wm + mi * 16 + lo) * 32 + hi * 16];
      a[mi] = cat8(*reinterpret_cast<const v8h*>(p),
                   *reinterpret_cast<const v8h*>(p + 8));
    }
    v16h b[2];
#pragma unroll
    for (int ni = 0; ni < 2; ++ni) {
      const f16* p = &Bs[cur][(wn + ni * 16 + lo) * 32 + hi * 16];
      b[ni] = *reinterpret_cast<const v16h*>(p);
    }
#pragma unroll
    for (int mi = 0; mi < 4; ++mi)
#pragma unroll
      for (int ni = 0; ni < 2; ++ni)
        acc[mi * 2 + ni] = wmma16(a[mi], b[ni], acc[mi * 2 + ni]);

    wait_cp();
    __syncthreads();
  }

#pragma unroll
  for (int mi = 0; mi < 4; ++mi) {
#pragma unroll
    for (int ni = 0; ni < 2; ++ni) {
      v8f c = acc[mi * 2 + ni];
#pragma unroll
      for (int r = 0; r < 8; ++r) {
        int row = m0 + wm + mi * 16 + r + 8 * hi;
        int col = n0 + wn + ni * 16 + lo;
        if (MODE == 0)      outh[(size_t)row * N + col] = (f16)c[r];
        else if (MODE == 1) outh[(size_t)col * M + row] = (f16)c[r];
        else                outf[(size_t)row * N + col] = c[r];
      }
    }
  }
}

// ---------------------------------------------------------------------------
// Flash attention, causal. Block = 4 waves = 4 query tiles (64 queries) of one
// head. Each 32-key K/V block is staged ONCE into LDS (async, double-buffered)
// and shared by all 4 waves. Per-wave causal skip uses a readfirstlane'd
// SCALAR condition so the compiler emits s_cbranch (EXEC stays all-ones inside
// -- required for WMMA, which is NOT skipped on EXEC==0).
// ---------------------------------------------------------------------------
__global__ __launch_bounds__(128, 1)
void flash_attn(const f16* __restrict__ Qp, const f16* __restrict__ Kp,
                const f16* __restrict__ VT, f16* __restrict__ Out) {
  __shared__ alignas(64) f16 Ks[2][32 * 64];   // [key 0..31][d 0..63]
  __shared__ alignas(64) f16 Vs[2][64 * 32];   // [d 0..63][key 0..31]
  __shared__ alignas(64) f16 Ps[4][16 * 32];   // per-wave P bounce

  const int tid  = threadIdx.x;
  const int lane = tid & 31;
  const int wid  = tid >> 5;
  const int lo   = lane & 15;
  const int hi   = lane >> 4;
  const int h    = blockIdx.y;
  const int qb0  = blockIdx.x * 64;        // block's first query
  const int q0   = qb0 + wid * 16;         // this wave's first query
  // Wave-uniform scalar copy of q0: forces scalar branches (no EXEC masking)
  const int q0s  = __builtin_amdgcn_readfirstlane(q0);
  f16* Pw = Ps[wid];

  // Stage K rows [k0,k0+32) x 64d and V^T rows [h*64,+64) x keys [k0,k0+32).
  auto stage = [&](int buf, int k0) {
#pragma unroll
    for (int it = 0; it < 2; ++it) {
      int id  = tid + it * 128;            // 0..255
      int rk  = id >> 3, ck = id & 7;      // K: 32 rows x 8 chunks
      cp16(Kp + (size_t)(k0 + rk) * Edim + h * HDim + ck * 8,
           &Ks[buf][rk * 64 + ck * 8]);
      int rv  = id >> 2, cv = id & 3;      // V: 64 rows x 4 chunks
      cp16(VT + (size_t)(h * HDim + rv) * Tdim + k0 + cv * 8,
           &Vs[buf][rv * 32 + cv * 8]);
    }
  };

  // Q fragments: two K-chunks of 32 over HD=64
  v16h qf[2];
#pragma unroll
  for (int c = 0; c < 2; ++c) {
    const f16* p = Qp + (size_t)(q0 + lo) * Edim + h * HDim + c * 32 + hi * 16;
    qf[c] = cat8(*reinterpret_cast<const v8h*>(p),
                 *reinterpret_cast<const v8h*>(p + 8));
  }

  v16h ones;
#pragma unroll
  for (int e = 0; e < 16; ++e) ones[e] = (f16)1.0f;

  v8f z = {};
  v8f Oacc[4];
#pragma unroll
  for (int j = 0; j < 4; ++j) Oacc[j] = z;
  float mrow[8], lrow[8];
#pragma unroll
  for (int r = 0; r < 8; ++r) { mrow[r] = -__builtin_inff(); lrow[r] = 0.f; }

  const int nkb = ((qb0 + 63) >> 5) + 1;   // covers keys up to block's last row
  stage(0, 0);
  wait_cp();
  __syncthreads();

  for (int kb = 0; kb < nkb; ++kb) {
    const int k0  = kb * 32;
    const int cur = kb & 1;
    if (kb + 1 < nkb) stage(cur ^ 1, (kb + 1) * 32);   // prefetch next K/V

    if (k0 <= q0s + 15) {   // SCALAR branch: wave inside its causal range
      // S = Q * K^T for 32 keys (B fragments from shared Ks)
      v8f S0 = z, S1 = z;
#pragma unroll
      for (int c = 0; c < 2; ++c) {
        v16h b0 = *reinterpret_cast<const v16h*>(&Ks[cur][lo * 64 + c * 32 + hi * 16]);
        v16h b1 = *reinterpret_cast<const v16h*>(&Ks[cur][(16 + lo) * 64 + c * 32 + hi * 16]);
        S0 = wmma16(qf[c], b0, S0);
        S1 = wmma16(qf[c], b1, S1);
      }

      // Masking only needed on diagonal blocks (scalar select of path)
      if (k0 + 31 > q0s)
        softmax_step<true >(S0, S1, q0, k0, lo, hi, mrow, lrow, Oacc, Pw);
      else
        softmax_step<false>(S0, S1, q0, k0, lo, hi, mrow, lrow, Oacc, Pw);

      // P in A-layout from this wave's private bounce region
      const f16* pp = &Pw[lo * 32 + hi * 16];
      v16h pf = cat8(*reinterpret_cast<const v8h*>(pp),
                     *reinterpret_cast<const v8h*>(pp + 8));

      // Row sums of P via WMMA against ones
      v8f rs = wmma16(pf, ones, z);
#pragma unroll
      for (int r = 0; r < 8; ++r) lrow[r] += rs[r];

      // O += P(16x32) * V(32x64) from shared Vs
#pragma unroll
      for (int j = 0; j < 4; ++j) {
        v16h vf = *reinterpret_cast<const v16h*>(&Vs[cur][(j * 16 + lo) * 32 + hi * 16]);
        Oacc[j] = wmma16(pf, vf, Oacc[j]);
      }
    }

    wait_cp();
    __syncthreads();
  }

  // Normalize and write f16 [T][E]
#pragma unroll
  for (int j = 0; j < 4; ++j) {
#pragma unroll
    for (int r = 0; r < 8; ++r) {
      int row = q0 + r + 8 * hi;
      int col = h * HDim + j * 16 + lo;
      Out[(size_t)row * Edim + col] = (f16)(Oacc[j][r] / lrow[r]);
    }
  }
}

// ---------------------------------------------------------------------------
extern "C" void kernel_launch(void* const* d_in, const int* in_sizes, int n_in,
                              void* d_out, int out_size, void* d_ws, size_t ws_size,
                              hipStream_t stream) {
  (void)in_sizes; (void)n_in; (void)out_size; (void)ws_size;
  const float* q  = (const float*)d_in[0];
  const float* k  = (const float*)d_in[1];
  const float* v  = (const float*)d_in[2];
  const float* Wq = (const float*)d_in[3];
  const float* Wk = (const float*)d_in[4];
  const float* Wv = (const float*)d_in[5];
  const float* Wo = (const float*)d_in[6];

  const size_t TE = (size_t)Tdim * Edim;   // 4M elems
  const size_t EE = (size_t)Edim * Edim;   // 1M elems
  f16* ws   = (f16*)d_ws;                  // needs 32M f16 = 64 MB
  f16* q16  = ws;
  f16* k16  = q16 + TE;
  f16* v16  = k16 + TE;
  f16* wq16 = v16 + TE;
  f16* wk16 = wq16 + EE;
  f16* wv16 = wk16 + EE;
  f16* wo16 = wv16 + EE;
  f16* qp   = wo16 + EE;
  f16* kp   = qp + TE;
  f16* vT   = kp + TE;
  f16* att  = vT + TE;

  const int nTE4 = (int)(TE / 4), nEE4 = (int)(EE / 4);
  cvt_f32_f16<<<nTE4 / 256, 256, 0, stream>>>(q,  q16,  nTE4);
  cvt_f32_f16<<<nTE4 / 256, 256, 0, stream>>>(k,  k16,  nTE4);
  cvt_f32_f16<<<nTE4 / 256, 256, 0, stream>>>(v,  v16,  nTE4);
  cvt_f32_f16<<<nEE4 / 256, 256, 0, stream>>>(Wq, wq16, nEE4);
  cvt_f32_f16<<<nEE4 / 256, 256, 0, stream>>>(Wk, wk16, nEE4);
  cvt_f32_f16<<<nEE4 / 256, 256, 0, stream>>>(Wv, wv16, nEE4);
  cvt_f32_f16<<<nEE4 / 256, 256, 0, stream>>>(Wo, wo16, nEE4);

  dim3 g(Edim / 128, Tdim / 128);   // (8, 32)
  gemm_xwT<0><<<g, 256, 0, stream>>>(q16, wq16, qp, nullptr, Tdim, Edim, Edim);
  gemm_xwT<0><<<g, 256, 0, stream>>>(k16, wk16, kp, nullptr, Tdim, Edim, Edim);
  gemm_xwT<1><<<g, 256, 0, stream>>>(v16, wv16, vT, nullptr, Tdim, Edim, Edim);

  flash_attn<<<dim3(Tdim / 64, Hn), 128, 0, stream>>>(qp, kp, vT, att);

  gemm_xwT<2><<<g, 256, 0, stream>>>(att, wo16, nullptr, (float*)d_out,
                                     Tdim, Edim, Edim);
}
